// SimpleGAT_63539746177578
// MI455X (gfx1250) — compile-verified, matrix operand order checked by reading, not run on hardware
//
#include <hip/hip_runtime.h>
#include <hip/hip_bf16.h>

typedef __attribute__((ext_vector_type(16))) _Float16 v16h;
typedef __attribute__((ext_vector_type(8)))  float    v8f;
typedef int v4i_ __attribute__((vector_size(16)));

#define BATCH 64
#define NNODE 1024
#define FIN   16
#define HC1   64    // 4 heads * 16
#define HC2   128   // 4 heads * 32
#define KNBR  8     // 7 knn + self

// ---- CDNA5 async global->LDS copy (ASYNCcnt path), with safe fallback ----
#if defined(__gfx1250__) && __has_builtin(__builtin_amdgcn_global_load_async_to_lds_b128)
#define HAVE_ASYNC_LDS 1
#else
#define HAVE_ASYNC_LDS 0
#endif

__device__ __forceinline__ void async_copy16(const void* g, void* l) {
#if HAVE_ASYNC_LDS
  // Builtin signature (from hipcc diagnostic): first param is
  // 'int __attribute__((vector_size(16))) __device__ *' (global AS, non-const).
  typedef __attribute__((address_space(1))) v4i_* gptr_t;
  typedef __attribute__((address_space(3))) v4i_* lptr_t;
  __builtin_amdgcn_global_load_async_to_lds_b128((gptr_t)g, (lptr_t)l,
                                                 /*offset=*/0, /*cpol=*/0);
#else
  *(uint4*)l = *(const uint4*)g;
#endif
}

__device__ __forceinline__ void async_wait0() {
#if HAVE_ASYNC_LDS
#if __has_builtin(__builtin_amdgcn_s_wait_asynccnt)
  __builtin_amdgcn_s_wait_asynccnt(0);
#else
  asm volatile("s_wait_asynccnt 0x0" ::: "memory");
#endif
#endif
}

// 16-bit A-matrix 16x32 per-lane K index (CDNA5 ISA 7.12.2):
// lanes 0-15 (half=0): VGPR0..3 -> K 0..7, VGPR4..7 -> K 16..23
// lanes 16-31 (half=1): K 8..15, 24..31
__device__ __forceinline__ int kmap(int j, int half) {
  return (j < 8 ? j : j + 8) + 8 * half;
}

// ---------------------------------------------------------------- kNN
__global__ __launch_bounds__(256) void knn_kernel(const float* __restrict__ pos,
                                                  int* __restrict__ idx) {
  __shared__ float sp[NNODE * 3];
  const int t = threadIdx.x;
  const int b = blockIdx.x >> 2;       // 4 blocks per batch
  const int tile = blockIdx.x & 3;
  const float* pb = pos + (long)b * NNODE * 3;
  // 3072 floats = 768 x 16B chunks, DMA'd straight into LDS
  for (int e = t; e < 768; e += 256) async_copy16(pb + e * 4, sp + e * 4);
  async_wait0();
  __syncthreads();

  const int i = tile * 256 + t;
  const float xi = sp[i * 3 + 0], yi = sp[i * 3 + 1], zi = sp[i * 3 + 2];
  float bd[7]; int bi[7];
#pragma unroll
  for (int k = 0; k < 7; ++k) { bd[k] = 3.4e38f; bi[k] = 0; }
  for (int j = 0; j < NNODE; ++j) {
    if (j == i) continue;
    float dx = xi - sp[j * 3 + 0];
    float dy = yi - sp[j * 3 + 1];
    float dz = zi - sp[j * 3 + 2];
    float d = dx * dx + dy * dy + dz * dz;
    if (d < bd[6]) {
      bd[6] = d; bi[6] = j;
#pragma unroll
      for (int q = 6; q > 0; --q) {         // bubble up; stable for ties
        if (bd[q] < bd[q - 1]) {
          float td = bd[q]; bd[q] = bd[q - 1]; bd[q - 1] = td;
          int ti = bi[q]; bi[q] = bi[q - 1]; bi[q - 1] = ti;
        }
      }
    }
  }
  int* op = idx + ((long)b * NNODE + i) * KNBR;
#pragma unroll
  for (int k = 0; k < 7; ++k) op[k] = b * NNODE + bi[k];
  op[7] = b * NNODE + i;  // self loop
}

// ---------------------------------------------------------------- layer-1 linear (WMMA)
// x[65536,16] (K padded to 32) @ W[16,64] + b  -> f16, for W_l and W_r.
// fp32->f16 conversion happens while staging, so this stays on the VGPR path.
__global__ __launch_bounds__(256) void linear1_kernel(
    const float* __restrict__ x,
    const float* __restrict__ Wl, const float* __restrict__ bl,
    const float* __restrict__ Wr, const float* __restrict__ br,
    _Float16* __restrict__ xl, _Float16* __restrict__ xr) {
  __shared__ _Float16 sX[128][36];        // rows x K (padded)
  __shared__ _Float16 sW[2][64][36];      // [mat][n][k] = W^T, K padded w/ zeros
  const int t = threadIdx.x;
  const long rowbase = (long)blockIdx.x * 128;

  { // stage X as f16
    int row = t >> 1, sub = t & 1;
    const float* xp = x + (rowbase + row) * FIN + sub * 8;
#pragma unroll
    for (int j = 0; j < 8; ++j) sX[row][sub * 8 + j] = (_Float16)xp[j];
  }
  for (int e = t; e < 128 * 20; e += 256) {      // zero K=16..35
    int row = e / 20, k = 16 + e % 20;
    sX[row][k] = (_Float16)0.f;
  }
  for (int e = t; e < 64 * 36; e += 256) {       // stage W transposed
    int n = e / 36, k = e % 36;
    sW[0][n][k] = (k < FIN) ? (_Float16)Wl[k * 64 + n] : (_Float16)0.f;
    sW[1][n][k] = (k < FIN) ? (_Float16)Wr[k * 64 + n] : (_Float16)0.f;
  }
  __syncthreads();

  const int wave = t >> 5, lane = t & 31;
  const int r = lane & 15, half = lane >> 4;
  v16h a;
#pragma unroll
  for (int j = 0; j < 16; ++j) a[j] = sX[wave * 16 + r][kmap(j, half)];

  const float* biases[2] = {bl, br};
  _Float16* outs[2] = {xl, xr};
#pragma unroll
  for (int m = 0; m < 2; ++m) {
#pragma unroll
    for (int nt = 0; nt < 4; ++nt) {
      v16h bf;
#pragma unroll
      for (int j = 0; j < 16; ++j) bf[j] = sW[m][nt * 16 + r][kmap(j, half)];
      v8f c = {};
      c = __builtin_amdgcn_wmma_f32_16x16x32_f16(false, a, false, bf,
                                                 (short)0, c, false, false);
      const int n = nt * 16 + r;
      const float bv = biases[m][n];
#pragma unroll
      for (int i = 0; i < 8; ++i) {            // C/D: VGPR i -> M = i + 8*half
        long row = rowbase + wave * 16 + i + 8 * half;
        outs[m][row * HC1 + n] = (_Float16)(c[i] + bv);
      }
    }
  }
}

// ---------------------------------------------------------------- layer-2 linear (WMMA)
// h1[65536,64] f16 @ W[64,128] + b -> f16, for W_l2 and W_r2.
__global__ __launch_bounds__(256) void linear2_kernel(
    const _Float16* __restrict__ h,
    const float* __restrict__ Wl, const float* __restrict__ bl,
    const float* __restrict__ Wr, const float* __restrict__ br,
    _Float16* __restrict__ xl, _Float16* __restrict__ xr) {
  __shared__ _Float16 sH[128][72];        // 144B row stride: 16B-aligned chunks
  __shared__ _Float16 sW[2][128][68];
  const int t = threadIdx.x;
  const long rowbase = (long)blockIdx.x * 128;

  // Async DMA: 128 rows x 8 chunks (16B each) straight into LDS, no VGPRs.
  for (int e = t; e < 128 * 8; e += 256) {
    int row = e >> 3, ch = e & 7;
    async_copy16(h + (rowbase + row) * HC1 + ch * 8, &sH[row][ch * 8]);
  }
  for (int e = t; e < 128 * 64; e += 256) {   // weights need transpose: VGPR path
    int n = e >> 6, k = e & 63;
    sW[0][n][k] = (_Float16)Wl[k * HC2 + n];
    sW[1][n][k] = (_Float16)Wr[k * HC2 + n];
  }
  async_wait0();
  __syncthreads();

  const int wave = t >> 5, lane = t & 31;
  const int r = lane & 15, half = lane >> 4;
  v16h a0, a1;
#pragma unroll
  for (int j = 0; j < 16; ++j) {
    a0[j] = sH[wave * 16 + r][kmap(j, half)];
    a1[j] = sH[wave * 16 + r][32 + kmap(j, half)];
  }
  const float* biases[2] = {bl, br};
  _Float16* outs[2] = {xl, xr};
#pragma unroll
  for (int m = 0; m < 2; ++m) {
#pragma unroll
    for (int nt = 0; nt < 8; ++nt) {
      v16h b0, b1;
#pragma unroll
      for (int j = 0; j < 16; ++j) {
        b0[j] = sW[m][nt * 16 + r][kmap(j, half)];
        b1[j] = sW[m][nt * 16 + r][32 + kmap(j, half)];
      }
      v8f c = {};
      c = __builtin_amdgcn_wmma_f32_16x16x32_f16(false, a0, false, b0,
                                                 (short)0, c, false, false);
      c = __builtin_amdgcn_wmma_f32_16x16x32_f16(false, a1, false, b1,
                                                 (short)0, c, false, false);
      const int n = nt * 16 + r;
      const float bv = biases[m][n];
#pragma unroll
      for (int i = 0; i < 8; ++i) {
        long row = rowbase + wave * 16 + i + 8 * half;
        outs[m][row * HC2 + n] = (_Float16)(c[i] + bv);
      }
    }
  }
}

// ---------------------------------------------------------------- GATv2 attention
// One thread per (node, head). Two-pass over 8 neighbors (rows stay L2/L0 hot).
template <int C, typename OUT>
__global__ __launch_bounds__(256) void attn_kernel(
    const _Float16* __restrict__ xl, const _Float16* __restrict__ xr,
    const int* __restrict__ idx, const float* __restrict__ att,
    const float* __restrict__ bias, OUT* __restrict__ out) {
  const int HC = 4 * C;
  const long g = (long)blockIdx.x * 256 + threadIdx.x;
  const long node = g >> 2;
  const int head = (int)(g & 3);

  float xrv[C], av[C];
  const _Float16* pr = xr + node * HC + head * C;
#pragma unroll
  for (int c = 0; c < C; ++c) xrv[c] = (float)pr[c];
#pragma unroll
  for (int c = 0; c < C; ++c) av[c] = att[head * C + c];

  const int* ip = idx + node * KNBR;
  float lg[KNBR], mx = -3.4e38f;
#pragma unroll
  for (int k = 0; k < KNBR; ++k) {
    long src = ip[k];
    const _Float16* pl = xl + src * HC + head * C;
    float s = 0.f;
#pragma unroll
    for (int c = 0; c < C; ++c) {
      float e = (float)pl[c] + xrv[c];
      e = (e > 0.f) ? e : 0.2f * e;   // leaky_relu, slope 0.2
      s += e * av[c];
    }
    lg[k] = s;
    mx = fmaxf(mx, s);
  }
  float den = 0.f;
#pragma unroll
  for (int k = 0; k < KNBR; ++k) { lg[k] = __expf(lg[k] - mx); den += lg[k]; }
  const float inv = 1.f / den;

  float acc[C];
#pragma unroll
  for (int c = 0; c < C; ++c) acc[c] = 0.f;
#pragma unroll
  for (int k = 0; k < KNBR; ++k) {
    const float w = lg[k] * inv;
    long src = ip[k];
    const _Float16* pl = xl + src * HC + head * C;
#pragma unroll
    for (int c = 0; c < C; ++c) acc[c] += w * (float)pl[c];
  }
  OUT* po = out + node * HC + head * C;
#pragma unroll
  for (int c = 0; c < C; ++c) {
    float v = acc[c] + bias[head * C + c];
    v = v > 0.f ? v : 0.f;            // relu
    po[c] = (OUT)v;
  }
}

// ---------------------------------------------------------------- mean pool
__global__ __launch_bounds__(128) void pool_kernel(const float* __restrict__ h2,
                                                   float* __restrict__ out) {
  const int b = blockIdx.x, c = threadIdx.x;
  const float* p = h2 + (long)b * NNODE * HC2 + c;
  float acc = 0.f;
  for (int n = 0; n < NNODE; ++n) acc += p[(long)n * HC2];
  out[b * HC2 + c] = acc * (1.f / (float)NNODE);
}

// ---------------------------------------------------------------- launch
extern "C" void kernel_launch(void* const* d_in, const int* in_sizes, int n_in,
                              void* d_out, int out_size, void* d_ws, size_t ws_size,
                              hipStream_t stream) {
  const float* x    = (const float*)d_in[0];
  const float* pos  = (const float*)d_in[1];
  const float* W_l1 = (const float*)d_in[2];
  const float* b_l1 = (const float*)d_in[3];
  const float* W_r1 = (const float*)d_in[4];
  const float* b_r1 = (const float*)d_in[5];
  const float* att1 = (const float*)d_in[6];
  const float* bias1= (const float*)d_in[7];
  const float* W_l2 = (const float*)d_in[8];
  const float* b_l2 = (const float*)d_in[9];
  const float* W_r2 = (const float*)d_in[10];
  const float* b_r2 = (const float*)d_in[11];
  const float* att2 = (const float*)d_in[12];
  const float* bias2= (const float*)d_in[13];

  const long M = (long)BATCH * NNODE;   // 65536
  char* ws = (char*)d_ws;
  auto carve = [&](size_t bytes) {
    char* p = ws;
    ws += (bytes + 255) & ~(size_t)255;
    return p;
  };
  int*      idxb = (int*)      carve(M * KNBR * sizeof(int));
  _Float16* xl1  = (_Float16*) carve(M * HC1 * sizeof(_Float16));
  _Float16* xr1  = (_Float16*) carve(M * HC1 * sizeof(_Float16));
  _Float16* h1   = (_Float16*) carve(M * HC1 * sizeof(_Float16));
  _Float16* xl2  = (_Float16*) carve(M * HC2 * sizeof(_Float16));
  _Float16* xr2  = (_Float16*) carve(M * HC2 * sizeof(_Float16));
  float*    h2   = (float*)    carve(M * HC2 * sizeof(float));

  knn_kernel<<<BATCH * 4, 256, 0, stream>>>(pos, idxb);
  linear1_kernel<<<M / 128, 256, 0, stream>>>(x, W_l1, b_l1, W_r1, b_r1, xl1, xr1);
  attn_kernel<16, _Float16><<<M * 4 / 256, 256, 0, stream>>>(xl1, xr1, idxb, att1, bias1, h1);
  linear2_kernel<<<M / 128, 256, 0, stream>>>(h1, W_l2, b_l2, W_r2, b_r2, xl2, xr2);
  attn_kernel<32, float><<<M * 4 / 256, 256, 0, stream>>>(xl2, xr2, idxb, att2, bias2, h2);
  pool_kernel<<<BATCH, 128, 0, stream>>>(h2, (float*)d_out);
}
